// LLaDAExpertGroup_21285857919732
// MI455X (gfx1250) — matrix operand order repaired and checked by reading, not compiled, not run on hardware
//
#include <hip/hip_runtime.h>
#include <hip/hip_bf16.h>

typedef __bf16 bf16;
typedef __attribute__((ext_vector_type(16))) __bf16 v16bf;
typedef __attribute__((ext_vector_type(8)))  float  v8f;

union FragU { v16bf v; unsigned u[8]; };

__device__ inline v8f zero8() {
  v8f z = {0.f,0.f,0.f,0.f,0.f,0.f,0.f,0.f};
  return z;
}

// ---- CDNA5 async global->LDS copy (ASYNCcnt path), 16B per lane ----------
__device__ inline void async_copy16(void* lds_dst, const void* gsrc) {
  unsigned ldsa = (unsigned)(size_t)lds_dst;        // low 32 bits = LDS offset
  asm volatile("global_load_async_to_lds_b128 %0, %1, off"
               :: "v"(ldsa), "v"(gsrc) : "memory");
}
__device__ inline void wait_async0() {
  asm volatile("s_wait_asynccnt 0" ::: "memory");
}

// ---------------------------------------------------------------------------
// Generic NT GEMM:  C[M,N] = A[M,K] * B[N,K]^T   (bf16 in, f32 accumulate)
// 256 threads = 8 waves; block tile 128(M) x 64(N); K step 32.
// Double-buffered LDS staged via GLOBAL_LOAD_ASYNC_TO_LDS_B128.
// Wave w computes rows [bm + 16w, +16), all 64 columns (4 accum frags).
// Fragment layouts follow CDNA5 ISA 7.12.2:
//   A 16x32 bf16: lane m(0-15) K{0..7,16..23}, lane m+16 K{8..15,24..31}
//   B 32x16 bf16: lane n(0-15) K{0..7,16..23}, lane n+16 K{8..15,24..31}
// EPI: 0 = f32 store; 2 = bf16 store; 3 = bf16(ADD + alpha*res); 4 = Cf += alpha*res
// ---------------------------------------------------------------------------
template<int EPI>
__global__ __launch_bounds__(256)
void gemm_bf16_nt(const bf16* __restrict__ A, const bf16* __restrict__ B,
                  const bf16* __restrict__ ADD, float* __restrict__ Cf,
                  bf16* __restrict__ Cb, int M, int N, int K, float alpha)
{
  __shared__ __align__(16) bf16 lsa[2][128 * 32];
  __shared__ __align__(16) bf16 lsb[2][64 * 32];
  const int tid  = threadIdx.x;
  const int wave = tid >> 5;
  const int lane = tid & 31;
  const int half = lane >> 4;
  const int l15  = lane & 15;
  const int bm = blockIdx.x * 128;
  const int bn = blockIdx.y * 64;

  // issue async loads of one K-slab into buffer bi
  auto issue = [&](int bi, int k0) {
    // A tile: 128 rows x 32 el = 512 16B-chunks; 2 per thread
    for (int c = tid; c < 512; c += 256) {
      int r = c >> 2, off = (c & 3) << 3;
      async_copy16(&lsa[bi][r * 32 + off], A + (long)(bm + r) * K + k0 + off);
    }
    // B tile: 64 rows x 32 el = 256 chunks; 1 per thread
    {
      int r = tid >> 2, off = (tid & 3) << 3;
      async_copy16(&lsb[bi][r * 32 + off], B + (long)(bn + r) * K + k0 + off);
    }
  };

  v8f acc[4];
#pragma unroll
  for (int i = 0; i < 4; ++i) acc[i] = zero8();

  const int nsteps = K >> 5;
  issue(0, 0);

  for (int s = 0; s < nsteps; ++s) {
    const int bi = s & 1;
    wait_async0();        // our async writes into buffer bi are done
    __syncthreads();      // everyone's are done; prior reads of bi^1 retired
    if (s + 1 < nsteps) issue(bi ^ 1, (s + 1) << 5);

    FragU af;
    const unsigned* pa = (const unsigned*)(&lsa[bi][(wave * 16 + l15) * 32]);
#pragma unroll
    for (int v = 0; v < 8; ++v) af.u[v] = pa[(v < 4 ? v : v + 4) + half * 4];

#pragma unroll
    for (int nb = 0; nb < 4; ++nb) {
      FragU bfr;
      const unsigned* pb = (const unsigned*)(&lsb[bi][(nb * 16 + l15) * 32]);
#pragma unroll
      for (int v = 0; v < 8; ++v) bfr.u[v] = pb[(v < 4 ? v : v + 4) + half * 4];
      acc[nb] = __builtin_amdgcn_wmma_f32_16x16x32_bf16(
          false, af.v, false, bfr.v, (short)0, acc[nb], false, false);
    }
  }

  // C layout: lanes 0-15 -> N=lane, M=vgpr r; lanes 16-31 -> N=lane-16, M=8+r
  const int crow = bm + wave * 16 + half * 8;
#pragma unroll
  for (int nb = 0; nb < 4; ++nb) {
    const int ccol = bn + nb * 16 + l15;
#pragma unroll
    for (int r = 0; r < 8; ++r) {
      long idx = (long)(crow + r) * N + ccol;
      float vres = acc[nb][r];
      if (EPI == 0)      Cf[idx] = vres;
      else if (EPI == 2) Cb[idx] = (bf16)vres;
      else if (EPI == 3) Cb[idx] = (bf16)((float)ADD[idx] + alpha * vres);
      else if (EPI == 4) Cf[idx] += alpha * vres;
    }
  }
}

// ---------------------------------------------------------------------------
// Fused per-batch "adapt attention":
//   adapt[s,:] = sum_t silu(clip(adapt_in[s,:] . adapt_out[t,:], +-5)) * adapt_in[t,:]
// Block = 128 threads, handles 64 s-rows; loops t in tiles of 64.
// Stage 1: 64x64 scores via WMMA (K = A = 128)  -> clip/silu -> bf16 P in LDS
// Stage 2: adapt(64x128) += P(64x64) @ V(64x128) via WMMA (K = 64)
// K/V tiles staged with GLOBAL_LOAD_ASYNC_TO_LDS_B128.
// ---------------------------------------------------------------------------
__global__ __launch_bounds__(128)
void adapt_attn_kernel(const bf16* __restrict__ qin,  // adapt_in  [B,S,A]
                       const bf16* __restrict__ kin,  // adapt_out [B,S,A]
                       float* __restrict__ outp,      // adapt     [B,S,A]
                       int S_, int A_)
{
  __shared__ __align__(16) bf16 lq[64 * 128];
  __shared__ __align__(16) bf16 lk[64 * 128];
  __shared__ __align__(16) bf16 lv[64 * 128];
  __shared__ __align__(16) bf16 lp[64 * 64];

  const int b  = blockIdx.y;
  const int s0 = blockIdx.x * 64;
  const bf16* Q  = qin + (long)b * S_ * A_;
  const bf16* Km = kin + (long)b * S_ * A_;

  const int tid  = threadIdx.x;
  const int wave = tid >> 5;
  const int lane = tid & 31;
  const int half = lane >> 4;
  const int l15  = lane & 15;

  // load Q tile once (64 rows x 128) via async copy
  for (int c = tid; c < 64 * 16; c += 128) {
    int r = c >> 4, off = (c & 15) << 3;
    async_copy16(lq + r * 128 + off, Q + (long)(s0 + r) * A_ + off);
  }

  v8f oacc[8];
#pragma unroll
  for (int i = 0; i < 8; ++i) oacc[i] = zero8();

  for (int t0 = 0; t0 < S_; t0 += 64) {
    __syncthreads();        // prior iteration's LDS reads retired
    for (int c = tid; c < 64 * 16; c += 128) {
      int r = c >> 4, off = (c & 15) << 3;
      async_copy16(lk + r * 128 + off, Km + (long)(t0 + r) * A_ + off);
      async_copy16(lv + r * 128 + off, Q  + (long)(t0 + r) * A_ + off);
    }
    wait_async0();          // also covers the initial Q tile on iter 0
    __syncthreads();

    // stage 1: scores
    v8f sacc[4];
#pragma unroll
    for (int i = 0; i < 4; ++i) sacc[i] = zero8();

#pragma unroll
    for (int kk = 0; kk < 128; kk += 32) {
      FragU af;
      const unsigned* pa = (const unsigned*)(lq + (wave * 16 + l15) * 128 + kk);
#pragma unroll
      for (int v = 0; v < 8; ++v) af.u[v] = pa[(v < 4 ? v : v + 4) + half * 4];
#pragma unroll
      for (int nb = 0; nb < 4; ++nb) {
        FragU bfr;
        const unsigned* pb = (const unsigned*)(lk + (nb * 16 + l15) * 128 + kk);
#pragma unroll
        for (int v = 0; v < 8; ++v) bfr.u[v] = pb[(v < 4 ? v : v + 4) + half * 4];
        sacc[nb] = __builtin_amdgcn_wmma_f32_16x16x32_bf16(
            false, af.v, false, bfr.v, (short)0, sacc[nb], false, false);
      }
    }

    // clip + silu, write P as bf16 to LDS
#pragma unroll
    for (int nb = 0; nb < 4; ++nb) {
#pragma unroll
      for (int r = 0; r < 8; ++r) {
        float s = sacc[nb][r];
        s = fminf(5.f, fmaxf(-5.f, s));
        s = s / (1.f + __expf(-s));
        int mr = wave * 16 + half * 8 + r;
        int nc = nb * 16 + l15;
        lp[mr * 64 + nc] = (bf16)s;
      }
    }
    __syncthreads();

    // stage 2: oacc += P @ V
#pragma unroll
    for (int kk2 = 0; kk2 < 64; kk2 += 32) {
      FragU pf;
      const unsigned* pp = (const unsigned*)(lp + (wave * 16 + l15) * 64 + kk2);
#pragma unroll
      for (int v = 0; v < 8; ++v) pf.u[v] = pp[(v < 4 ? v : v + 4) + half * 4];
#pragma unroll
      for (int nb = 0; nb < 8; ++nb) {
        FragU vf;
        const int col = nb * 16 + l15;
#pragma unroll
        for (int v = 0; v < 8; ++v) {
          int k = kk2 + 2 * v + (v < 4 ? 0 : 8) + 8 * half;     // V is K(t)-major: pack 2 rows
          union { bf16 h[2]; unsigned u; } pk;
          pk.h[0] = lv[(long)k * 128 + col];
          pk.h[1] = lv[(long)(k + 1) * 128 + col];
          vf.u[v] = pk.u;
        }
        oacc[nb] = __builtin_amdgcn_wmma_f32_16x16x32_bf16(
            false, pf.v, false, vf.v, (short)0, oacc[nb], false, false);
      }
    }
  }

  const int orow = s0 + wave * 16 + half * 8;
#pragma unroll
  for (int nb = 0; nb < 8; ++nb) {
#pragma unroll
    for (int r = 0; r < 8; ++r) {
      outp[(long)b * S_ * A_ + (long)(orow + r) * A_ + nb * 16 + l15] = oacc[nb][r];
    }
  }
}

// ---------------------------------------------------------------------------
// Small helper kernels
// ---------------------------------------------------------------------------
__global__ void cvt_f32_bf16(const float* __restrict__ s, bf16* __restrict__ d, long n)
{
  long i = (long)blockIdx.x * blockDim.x + threadIdx.x;
  long stride = (long)gridDim.x * blockDim.x;
  for (; i < n; i += stride) d[i] = (bf16)s[i];
}

__global__ void silu_mul_kernel(const bf16* __restrict__ g, const bf16* __restrict__ u,
                                bf16* __restrict__ h, long n)
{
  long i = (long)blockIdx.x * blockDim.x + threadIdx.x;
  long stride = (long)gridDim.x * blockDim.x;
  for (; i < n; i += stride) {
    float gf = (float)g[i];
    float s = gf / (1.f + __expf(-gf));
    h[i] = (bf16)(s * (float)u[i]);
  }
}

// M[d,a] = sum_h Wout[d,h] * Wexp[h,a]  (composed expert tail), store bf16
__global__ void compose_M_kernel(const float* __restrict__ Wout, const float* __restrict__ Wexp,
                                 bf16* __restrict__ Mb, int D_, int H_, int A_)
{
  int idx = blockIdx.x * 256 + threadIdx.x;
  if (idx >= D_ * A_) return;
  int d = idx / A_, a = idx % A_;
  float s = 0.f;
  for (int h = 0; h < H_; ++h) s += Wout[(long)d * H_ + h] * Wexp[(long)h * A_ + a];
  Mb[idx] = (bf16)s;
}

__device__ inline float block_sum_128(float v, float* red)
{
  int t = threadIdx.x;
  red[t] = v; __syncthreads();
  for (int s = 64; s > 0; s >>= 1) {
    if (t < s) red[t] += red[t + s];
    __syncthreads();
  }
  float r = red[0];
  __syncthreads();
  return r;
}

// one block (128 threads) per row of [rows, 128]; out = bf16(LN(src)*g + b)
__global__ __launch_bounds__(128)
void ln_kernel(const float* __restrict__ src, const float* __restrict__ g,
               const float* __restrict__ b, bf16* __restrict__ dst, int A_)
{
  __shared__ float red[128];
  long row = blockIdx.x;
  float x = src[row * A_ + threadIdx.x];
  float mean = block_sum_128(x, red) / (float)A_;
  float dx = x - mean;
  float var = block_sum_128(dx * dx, red) / (float)A_;
  float y = dx * rsqrtf(var + 1e-5f) * g[threadIdx.x] + b[threadIdx.x];
  dst[row * A_ + threadIdx.x] = (bf16)y;
}

// sel[n] = last expert with weight > 0, else -1
__global__ void sel_kernel(const float* __restrict__ ew, int* __restrict__ sel, long n, int E_)
{
  long i = (long)blockIdx.x * blockDim.x + threadIdx.x;
  long stride = (long)gridDim.x * blockDim.x;
  for (; i < n; i += stride) {
    int s = -1;
    for (int e = 0; e < E_; ++e)
      if (ew[i * E_ + e] > 0.f) s = e;
    sel[i] = s;
  }
}

// per-token expert adapter: h1 = LN_{Ag[sel],Ab[sel]}(pre . A_w[sel]^T); 0 if sel<0
__global__ __launch_bounds__(128)
void expert_h1_kernel(const float* __restrict__ pre, const float* __restrict__ Aw,
                      const float* __restrict__ Ag, const float* __restrict__ Ab,
                      const int* __restrict__ sel, bf16* __restrict__ h1, int A_)
{
  __shared__ float sp[128];
  __shared__ float red[128];
  long n = blockIdx.x;
  int t = threadIdx.x;
  int e = sel[n];
  if (e < 0) { h1[n * A_ + t] = (bf16)0.f; return; }  // uniform per block
  sp[t] = pre[n * A_ + t];
  __syncthreads();
  const float* W = Aw + (long)e * A_ * A_ + (long)t * A_;
  float acc = 0.f;
  for (int a = 0; a < A_; ++a) acc += sp[a] * W[a];
  float mean = block_sum_128(acc, red) / (float)A_;
  float dx = acc - mean;
  float var = block_sum_128(dx * dx, red) / (float)A_;
  float y = dx * rsqrtf(var + 1e-5f) * Ag[(long)e * A_ + t] + Ab[(long)e * A_ + t];
  h1[n * A_ + t] = (bf16)y;
}

// ---------------------------------------------------------------------------
extern "C" void kernel_launch(void* const* d_in, const int* in_sizes, int n_in,
                              void* d_out, int out_size, void* d_ws, size_t ws_size,
                              hipStream_t stream)
{
  (void)in_sizes; (void)n_in; (void)out_size; (void)ws_size;
  const int Bb = 4, S = 2048, D = 1024, Hh = 2048, Aa = 128, Ee = 8;
  const long N = (long)Bb * S;   // 8192 tokens

  const float* x     = (const float*)d_in[0];
  const float* ew    = (const float*)d_in[1];
  const float* Wup   = (const float*)d_in[2];
  const float* Wgate = (const float*)d_in[3];
  const float* Wdown = (const float*)d_in[4];
  const float* Wpre  = (const float*)d_in[5];
  const float* Wpost = (const float*)d_in[6];
  const float* lng   = (const float*)d_in[7];
  const float* lnb   = (const float*)d_in[8];
  const float* Wap   = (const float*)d_in[9];
  const float* Aw    = (const float*)d_in[10];
  const float* Ag    = (const float*)d_in[11];
  const float* Ab    = (const float*)d_in[12];
  const float* Wexp  = (const float*)d_in[13];
  const float* Wout  = (const float*)d_in[14];
  float* out = (float*)d_out;

  char* p = (char*)d_ws;
  auto carve = [&](size_t bytes) -> char* {
    char* q = p;
    p += (bytes + 255) & ~(size_t)255;
    return q;
  };
  bf16* xb      = (bf16*)carve(N * D * 2);
  bf16* wub     = (bf16*)carve((long)Hh * D * 2);
  bf16* wgb     = (bf16*)carve((long)Hh * D * 2);
  bf16* wdb     = (bf16*)carve((long)D * Hh * 2);
  bf16* wpreb   = (bf16*)carve((long)Aa * D * 2);
  bf16* wpostb  = (bf16*)carve((long)Aa * Hh * 2);
  bf16* wapb    = (bf16*)carve((long)Hh * Aa * 2);
  bf16* Mb      = (bf16*)carve((long)D * Aa * 2);
  float* pre_f  = (float*)carve(N * Aa * 4);
  float* tmp_f  = (float*)carve(N * Aa * 4);
  float* adapt_f= (float*)carve(N * Aa * 4);
  bf16* ain_b   = (bf16*)carve(N * Aa * 2);
  bf16* aout_b  = (bf16*)carve(N * Aa * 2);
  bf16* adapt_b = (bf16*)carve(N * Aa * 2);
  bf16* h1_b    = (bf16*)carve(N * Aa * 2);
  int*  sel     = (int*)carve(N * 4);
  bf16* up_b    = (bf16*)carve(N * Hh * 2);
  bf16* gate_b  = (bf16*)carve(N * Hh * 2);   // reused as hplus after silu_mul
  bf16* hid_b   = (bf16*)carve(N * Hh * 2);
  bf16* hplus_b = gate_b;

  auto cvt = [&](const float* s, bf16* d, long n) {
    int blocks = (int)((n + 255) / 256); if (blocks > 4096) blocks = 4096;
    cvt_f32_bf16<<<blocks, 256, 0, stream>>>(s, d, n);
  };

  // 1) precision casts
  cvt(x, xb, N * D);
  cvt(Wup, wub, (long)Hh * D);
  cvt(Wgate, wgb, (long)Hh * D);
  cvt(Wdown, wdb, (long)D * Hh);
  cvt(Wpre, wpreb, (long)Aa * D);
  cvt(Wpost, wpostb, (long)Aa * Hh);
  cvt(Wap, wapb, (long)Hh * Aa);

  // 2) compose expert tail M = Wout @ Wexp  (D x A)
  compose_M_kernel<<<(D * Aa + 255) / 256, 256, 0, stream>>>(Wout, Wexp, Mb, D, Hh, Aa);

  // 3) pre = x @ Wpre^T   [N, A]
  gemm_bf16_nt<0><<<dim3(N / 128, Aa / 64), 256, 0, stream>>>(
      xb, wpreb, nullptr, pre_f, nullptr, (int)N, Aa, D, 1.f);

  // 4) adapt_in = LN(pre)
  ln_kernel<<<(int)N, 128, 0, stream>>>(pre_f, lng, lnb, ain_b, Aa);

  // 5) up / gate GEMMs  [N, H]
  gemm_bf16_nt<2><<<dim3(N / 128, Hh / 64), 256, 0, stream>>>(
      xb, wub, nullptr, nullptr, up_b, (int)N, Hh, D, 1.f);
  gemm_bf16_nt<2><<<dim3(N / 128, Hh / 64), 256, 0, stream>>>(
      xb, wgb, nullptr, nullptr, gate_b, (int)N, Hh, D, 1.f);

  // 6) hidden = silu(gate) * up
  silu_mul_kernel<<<4096, 256, 0, stream>>>(gate_b, up_b, hid_b, N * Hh);

  // 7) tmp = hidden @ Wpost^T  [N, A];  8) adapt_out = LN(tmp)
  gemm_bf16_nt<0><<<dim3(N / 128, Aa / 64), 256, 0, stream>>>(
      hid_b, wpostb, nullptr, tmp_f, nullptr, (int)N, Aa, Hh, 1.f);
  ln_kernel<<<(int)N, 128, 0, stream>>>(tmp_f, lng, lnb, aout_b, Aa);

  // 9) fused adapt attention (per batch)
  adapt_attn_kernel<<<dim3(S / 64, Bb), 128, 0, stream>>>(ain_b, aout_b, adapt_f, S, Aa);

  // 10) adapt -> bf16
  cvt(adapt_f, adapt_b, N * Aa);

  // 11) hplus = bf16(hidden + 0.1 * adapt @ Wap^T)  [N, H]
  gemm_bf16_nt<3><<<dim3(N / 128, Hh / 64), 256, 0, stream>>>(
      adapt_b, wapb, hid_b, nullptr, hplus_b, (int)N, Hh, Aa, 0.1f);

  // 12) shared = hplus @ Wdown^T  -> d_out  [N, D]
  gemm_bf16_nt<0><<<dim3(N / 128, D / 64), 256, 0, stream>>>(
      hplus_b, wdb, nullptr, out, nullptr, (int)N, D, Hh, 1.f);

  // 13) expert selection (last positive expert wins)
  sel_kernel<<<(int)((N + 255) / 256), 256, 0, stream>>>(ew, sel, N, Ee);

  // 14) per-token expert adapter h1
  expert_h1_kernel<<<(int)N, 128, 0, stream>>>(pre_f, Aw, Ag, Ab, sel, h1_b, Aa);

  // 15) d_out += 0.1 * h1 @ M^T  [N, D]
  gemm_bf16_nt<4><<<dim3(N / 128, D / 64), 256, 0, stream>>>(
      h1_b, Mb, nullptr, out, nullptr, (int)N, D, Aa, 0.1f);
}
